// SFT_6777458393262
// MI455X (gfx1250) — matrix operand order repaired and checked by reading, not compiled, not run on hardware
//
#include <hip/hip_runtime.h>

#define B_   16
#define L1_  128
#define L2D_ 64
#define D_   512
#define H_   256
#define G4_  1024   // 4*H
#define FD_  1536   // 3*D
#define SEM_ 300

typedef __bf16 bf16_t;
typedef __attribute__((ext_vector_type(16))) __bf16 v16bf;
typedef __attribute__((ext_vector_type(8)))  __bf16 v8bf;
typedef __attribute__((ext_vector_type(8)))  float  v8f;

static __device__ __forceinline__ v8f wmma_bf16(v16bf a, v16bf b, v8f c) {
  // D = A(16x32 bf16) * B(32x16 bf16) + C(16x16 f32)
  return __builtin_amdgcn_wmma_f32_16x16x32_bf16(false, a, false, b, (short)0, c,
                                                 false, false);
}

// A fragment (16x32 bf16). Lane&15 = row M; 16 contiguous K values
// [kh, kh+16) with kh = 16*(lane>>4).
static __device__ __forceinline__ v16bf load_afrag(const bf16_t* base, int stride,
                                                   int m, int k0, int lane) {
  const int row = m + (lane & 15);
  const int kh  = k0 + ((lane >> 4) << 4);
  v8bf lo = *(const v8bf*)(base + (size_t)row * stride + kh);
  v8bf hi = *(const v8bf*)(base + (size_t)row * stride + kh + 8);
  return __builtin_shufflevector(lo, hi, 0, 1, 2, 3, 4, 5, 6, 7,
                                 8, 9, 10, 11, 12, 13, 14, 15);
}

// B fragment (32x16 bf16). Lane&15 = column N; 16 contiguous K values starting
// at 16*(lane>>4). For out = X @ W^T with W row-major [N][K] this is just a
// contiguous 32B read of a W row.
static __device__ __forceinline__ v16bf load_bfrag(const bf16_t* W, int stride,
                                                   int n, int k0, int lane) {
  const int col = n + (lane & 15);
  const int kb  = k0 + ((lane >> 4) << 4);
  return *(const v16bf*)(W + (size_t)col * stride + kb);
}

static __device__ __forceinline__ float sigf(float x) {
  return 1.0f / (1.0f + expf(-x));
}

// ---------------------------------------------------------------- converts
__global__ void cvt_bf16_kernel(const float* __restrict__ src,
                                bf16_t* __restrict__ dst, int n) {
  int i = blockIdx.x * blockDim.x + threadIdx.x;
  if (i < n) dst[i] = (bf16_t)src[i];
}

// scatter visual (rows of 512) into features buffer (row stride 1536, offset 0)
__global__ void cvt_vis_feat_kernel(const float* __restrict__ vis,
                                    bf16_t* __restrict__ feat, int n) {
  int i = blockIdx.x * blockDim.x + threadIdx.x;
  if (i < n) {
    int r = i >> 9;
    int c = i & 511;
    feat[(size_t)r * FD_ + c] = (bf16_t)vis[i];
  }
}

// ------------------------------------------------------------- WMMA GEMM
// out[M,N] = A[M,K](bf16) @ W[N,K](bf16)^T + bias0 + bias1   (f32 out)
// block = 256 threads = 8 waves; block tile 64(M) x 128(N); wave tile 32x32.
__global__ __launch_bounds__(256) void gemm_bf16_kernel(
    const bf16_t* __restrict__ A, const bf16_t* __restrict__ W,
    const float* __restrict__ bias0, const float* __restrict__ bias1,
    float* __restrict__ out, int M, int N, int K) {
  const int lane = threadIdx.x & 31;
  const int wave = threadIdx.x >> 5;
  const int m0 = blockIdx.y * 64 + (wave >> 2) * 32;
  const int n0 = blockIdx.x * 128 + (wave & 3) * 32;
  v8f acc00 = {}, acc01 = {}, acc10 = {}, acc11 = {};
  for (int k0 = 0; k0 < K; k0 += 32) {
    v16bf a0 = load_afrag(A, K, m0,      k0, lane);
    v16bf a1 = load_afrag(A, K, m0 + 16, k0, lane);
    v16bf b0 = load_bfrag(W, K, n0,      k0, lane);
    v16bf b1 = load_bfrag(W, K, n0 + 16, k0, lane);
    acc00 = wmma_bf16(a0, b0, acc00);
    acc01 = wmma_bf16(a0, b1, acc01);
    acc10 = wmma_bf16(a1, b0, acc10);
    acc11 = wmma_bf16(a1, b1, acc11);
  }
  const int lr    = lane & 15;
  const int mhalf = (lane >> 4) << 3;
  v8f accs[2][2] = {{acc00, acc01}, {acc10, acc11}};
  for (int tm = 0; tm < 2; ++tm) {
    for (int tn = 0; tn < 2; ++tn) {
      const int n = n0 + tn * 16 + lr;
      float bsum = 0.0f;
      if (bias0) bsum += bias0[n];
      if (bias1) bsum += bias1[n];
      const int mb = m0 + tm * 16 + mhalf;
      v8f a = accs[tm][tn];
#pragma unroll
      for (int j = 0; j < 8; ++j)
        out[(size_t)(mb + j) * N + n] = a[j] + bsum;
    }
  }
}

// ------------------------------------------------- additive attention pool
// One block per (b,l).  pooled = Wf + b + softmax(scores) @ U  with
// scores[o] = sum_d tanh(Wf[d]+U[o][d]+b[d]) * ww[d] + wb.
// Result written as bf16 directly into the features buffer at featOff.
__global__ __launch_bounds__(256) void attn_kernel(
    const float* __restrict__ Wf, const float* __restrict__ U,
    const float* __restrict__ battn, const float* __restrict__ ww,
    const float* __restrict__ wb, bf16_t* __restrict__ feat, int featOff) {
  const int bl   = blockIdx.x;          // b*128 + l
  const int b    = bl >> 7;
  const int tid  = threadIdx.x;
  const int lane = tid & 31;
  const int wave = tid >> 5;
  __shared__ float sc[L2D_];
  __shared__ float sinv;
  const float* wfrow = Wf + (size_t)bl * D_;
  const float* ub    = U + (size_t)b * L2D_ * D_;
  for (int oi = 0; oi < 8; ++oi) {
    const int o = wave * 8 + oi;
    const float* urow = ub + (size_t)o * D_;
    float s = 0.0f;
    for (int d = lane; d < D_; d += 32)
      s += tanhf(wfrow[d] + urow[d] + battn[d]) * ww[d];
#pragma unroll
    for (int off = 16; off > 0; off >>= 1) s += __shfl_xor(s, off, 32);
    if (lane == 0) sc[o] = s + wb[0];
  }
  __syncthreads();
  if (tid == 0) {
    float mx = sc[0];
    for (int o = 1; o < L2D_; ++o) mx = fmaxf(mx, sc[o]);
    float sum = 0.0f;
    for (int o = 0; o < L2D_; ++o) { float e = expf(sc[o] - mx); sc[o] = e; sum += e; }
    sinv = 1.0f / sum;
  }
  __syncthreads();
  const float inv = sinv;
  for (int d = tid; d < D_; d += 256) {
    float acc = wfrow[d] + battn[d];
    for (int o = 0; o < L2D_; ++o) acc += sc[o] * inv * ub[(size_t)o * D_ + d];
    feat[(size_t)bl * FD_ + featOff + d] = (bf16_t)acc;
  }
}

// --------------------------------------------------------- BiLSTM recurrence
// grid=2 (dir), block=1024 (32 waves).  Each wave owns 2 N-tiles of the 1024
// gate outputs.  h lives in LDS (bf16), gates staged in LDS (f32), cell state
// in registers.  Whh streamed each step (stays hot in L2).
__global__ __launch_bounds__(1024) void lstm_kernel(
    const float* __restrict__ pre_f, const float* __restrict__ pre_b,
    const bf16_t* __restrict__ whh_f, const bf16_t* __restrict__ whh_b,
    float* __restrict__ out) {
  const int dir   = blockIdx.x;
  const float*  pre = dir ? pre_b : pre_f;
  const bf16_t* whh = dir ? whh_b : whh_f;
  const int tid  = threadIdx.x;
  const int lane = tid & 31;
  const int wave = tid >> 5;

  __shared__ __align__(32) bf16_t hsh[B_ * H_];   //  8 KB
  __shared__ float gsh[B_ * G4_];                 // 64 KB

  for (int i = tid; i < B_ * H_; i += 1024) hsh[i] = (bf16_t)0.0f;
  float creg[4] = {0.0f, 0.0f, 0.0f, 0.0f};
  const int n0a = (wave * 2) * 16;
  const int n0b = n0a + 16;
  __syncthreads();

  for (int t = 0; t < L1_; ++t) {
    const int tm = dir ? (L1_ - 1 - t) : t;
    if (t + 1 < L1_) {  // prefetch next step's pre-activations into cache
      const int tnx = dir ? (L1_ - 2 - t) : (t + 1);
      const int pm  = tid >> 6;
      const int pn  = (tid & 63) << 4;
      __builtin_prefetch(&pre[((size_t)pm * L1_ + tnx) * G4_ + pn], 0, 3);
    }
    v8f acc0 = {}, acc1 = {};
#pragma unroll
    for (int kk = 0; kk < 8; ++kk) {
      v16bf a  = load_afrag(hsh, H_, 0, kk * 32, lane);   // LDS -> ds loads
      v16bf b0 = load_bfrag(whh, H_, n0a, kk * 32, lane);
      v16bf b1 = load_bfrag(whh, H_, n0b, kk * 32, lane);
      acc0 = wmma_bf16(a, b0, acc0);
      acc1 = wmma_bf16(a, b1, acc1);
    }
    const int lr    = lane & 15;
    const int mhalf = (lane >> 4) << 3;
    {
      const int n = n0a + lr;
#pragma unroll
      for (int j = 0; j < 8; ++j) {
        const int m = mhalf + j;
        gsh[m * G4_ + n] = acc0[j] + pre[((size_t)m * L1_ + tm) * G4_ + n];
      }
    }
    {
      const int n = n0b + lr;
#pragma unroll
      for (int j = 0; j < 8; ++j) {
        const int m = mhalf + j;
        gsh[m * G4_ + n] = acc1[j] + pre[((size_t)m * L1_ + tm) * G4_ + n];
      }
    }
    __syncthreads();
#pragma unroll
    for (int q = 0; q < 4; ++q) {
      const int e = tid + q * 1024;
      const int m = e >> 8;
      const int j = e & 255;
      const float ig = gsh[m * G4_ + j];
      const float fg = gsh[m * G4_ + j + 256];
      const float gg = gsh[m * G4_ + j + 512];
      const float og = gsh[m * G4_ + j + 768];
      const float c  = sigf(fg) * creg[q] + sigf(ig) * tanhf(gg);
      creg[q] = c;
      const float h  = sigf(og) * tanhf(c);
      out[((size_t)m * L1_ + tm) * (2 * H_) + dir * H_ + j] = h;
      hsh[m * H_ + j] = (bf16_t)h;
    }
    __syncthreads();
  }
}

// --------------------------------------------- max over time + semantics FC
__global__ __launch_bounds__(512) void maxfc_kernel(
    const float* __restrict__ out, const float* __restrict__ fcw,
    const float* __restrict__ fcb, float* __restrict__ sem) {
  const int b   = blockIdx.x;
  const int tid = threadIdx.x;  // 512
  __shared__ float vsh[512];
  const float* ob = out + (size_t)b * L1_ * (2 * H_);
  float mx = ob[tid];
  for (int t = 1; t < L1_; ++t) mx = fmaxf(mx, ob[(size_t)t * (2 * H_) + tid]);
  vsh[tid] = mx;
  __syncthreads();
  if (tid < SEM_) {
    const float* wr = fcw + (size_t)tid * (2 * H_);
    float acc = fcb[tid];
    for (int d = 0; d < 2 * H_; ++d) acc += vsh[d] * wr[d];
    sem[(size_t)b * SEM_ + tid] = acc;
  }
}

// ---------------------------------------------------------------- launcher
extern "C" void kernel_launch(void* const* d_in, const int* in_sizes, int n_in,
                              void* d_out, int out_size, void* d_ws, size_t ws_size,
                              hipStream_t stream) {
  const float* visual  = (const float*)d_in[0];
  const float* subject = (const float*)d_in[1];
  const float* predict = (const float*)d_in[2];
  const float* Ws_w  = (const float*)d_in[3];
  const float* Ws_b  = (const float*)d_in[4];
  const float* Uo_w  = (const float*)d_in[5];
  const float* Uo_b  = (const float*)d_in[6];
  const float* Um_w  = (const float*)d_in[7];
  const float* Um_b  = (const float*)d_in[8];
  const float* bo    = (const float*)d_in[9];
  const float* bm    = (const float*)d_in[10];
  const float* wo_w  = (const float*)d_in[11];
  const float* wo_b  = (const float*)d_in[12];
  const float* wm_w  = (const float*)d_in[13];
  const float* wm_b  = (const float*)d_in[14];
  const float* Wih_f = (const float*)d_in[15];
  const float* Whh_f = (const float*)d_in[16];
  const float* bih_f = (const float*)d_in[17];
  const float* bhh_f = (const float*)d_in[18];
  const float* Wih_b = (const float*)d_in[19];
  const float* Whh_b = (const float*)d_in[20];
  const float* bih_b = (const float*)d_in[21];
  const float* bhh_b = (const float*)d_in[22];
  const float* fc_w  = (const float*)d_in[23];
  const float* fc_b  = (const float*)d_in[24];

  float* outp = (float*)d_out;                            // (16,128,512)
  float* semp = outp + (size_t)B_ * L1_ * (2 * H_);       // (16,300)

  char* w = (char*)d_ws;
  auto carve = [&](size_t bytes) -> void* {
    void* p = (void*)w;
    w += (bytes + 255) & ~(size_t)255;
    return p;
  };
  bf16_t* vis16  = (bf16_t*)carve((size_t)B_ * L1_ * D_ * 2);
  bf16_t* sub16  = (bf16_t*)carve((size_t)B_ * L2D_ * D_ * 2);
  bf16_t* prd16  = (bf16_t*)carve((size_t)B_ * L2D_ * D_ * 2);
  bf16_t* Ws16   = (bf16_t*)carve((size_t)D_ * D_ * 2);
  bf16_t* Uo16   = (bf16_t*)carve((size_t)D_ * D_ * 2);
  bf16_t* Um16   = (bf16_t*)carve((size_t)D_ * D_ * 2);
  bf16_t* Wihf16 = (bf16_t*)carve((size_t)G4_ * FD_ * 2);
  bf16_t* Wihb16 = (bf16_t*)carve((size_t)G4_ * FD_ * 2);
  bf16_t* Whhf16 = (bf16_t*)carve((size_t)G4_ * H_ * 2);
  bf16_t* Whhb16 = (bf16_t*)carve((size_t)G4_ * H_ * 2);
  bf16_t* feat16 = (bf16_t*)carve((size_t)B_ * L1_ * FD_ * 2);
  float*  Wf32   = (float*)carve((size_t)B_ * L1_ * D_ * 4);
  float*  Uo32   = (float*)carve((size_t)B_ * L2D_ * D_ * 4);
  float*  Um32   = (float*)carve((size_t)B_ * L2D_ * D_ * 4);
  float*  pre_f  = (float*)carve((size_t)B_ * L1_ * G4_ * 4);
  float*  pre_b  = (float*)carve((size_t)B_ * L1_ * G4_ * 4);

  auto cvt = [&](const float* s, bf16_t* d, int n) {
    cvt_bf16_kernel<<<(n + 255) / 256, 256, 0, stream>>>(s, d, n);
  };
  cvt(visual,  vis16,  B_ * L1_ * D_);
  cvt(subject, sub16,  B_ * L2D_ * D_);
  cvt(predict, prd16,  B_ * L2D_ * D_);
  cvt(Ws_w,  Ws16,   D_ * D_);
  cvt(Uo_w,  Uo16,   D_ * D_);
  cvt(Um_w,  Um16,   D_ * D_);
  cvt(Wih_f, Wihf16, G4_ * FD_);
  cvt(Wih_b, Wihb16, G4_ * FD_);
  cvt(Whh_f, Whhf16, G4_ * H_);
  cvt(Whh_b, Whhb16, G4_ * H_);
  {
    int n = B_ * L1_ * D_;
    cvt_vis_feat_kernel<<<(n + 255) / 256, 256, 0, stream>>>(visual, feat16, n);
  }

  auto gemm = [&](const bf16_t* A, const bf16_t* Wm, const float* b0,
                  const float* b1, float* O, int M, int N, int K) {
    dim3 grid(N / 128, M / 64);
    gemm_bf16_kernel<<<grid, 256, 0, stream>>>(A, Wm, b0, b1, O, M, N, K);
  };
  gemm(vis16, Ws16, Ws_b, nullptr, Wf32, B_ * L1_,  D_, D_);
  gemm(sub16, Uo16, Uo_b, nullptr, Uo32, B_ * L2D_, D_, D_);
  gemm(prd16, Um16, Um_b, nullptr, Um32, B_ * L2D_, D_, D_);

  // features layout: [visual | attn_motion | attn_objects]
  attn_kernel<<<B_ * L1_, 256, 0, stream>>>(Wf32, Um32, bm, wm_w, wm_b, feat16, D_);
  attn_kernel<<<B_ * L1_, 256, 0, stream>>>(Wf32, Uo32, bo, wo_w, wo_b, feat16, 2 * D_);

  gemm(feat16, Wihf16, bih_f, bhh_f, pre_f, B_ * L1_, G4_, FD_);
  gemm(feat16, Wihb16, bih_b, bhh_b, pre_b, B_ * L1_, G4_, FD_);

  lstm_kernel<<<2, 1024, 0, stream>>>(pre_f, pre_b, Whhf16, Whhb16, outp);
  maxfc_kernel<<<B_, 512, 0, stream>>>(outp, fc_w, fc_b, semp);

  (void)in_sizes; (void)n_in; (void)out_size; (void)ws_size;
}
